// GraphAttentionLayer_86457691669078
// MI455X (gfx1250) — matrix operand order repaired and checked by reading, not compiled, not run on hardware
//
#include <hip/hip_runtime.h>
#include <hip/hip_bf16.h>
#include <stdint.h>

#define B_    4
#define N_    2048
#define DIN_  256
#define DOUT_ 256
#define H_    4
#define ALPHA_ 0.2f
#define LAM_   0.2f

#ifndef USE_ASYNC_LDS
#define USE_ASYNC_LDS 1
#endif

#if USE_ASYNC_LDS
#if __has_builtin(__builtin_amdgcn_global_load_async_to_lds_b32)
#define HAVE_ASYNC_B32 1
#endif
#endif

typedef __attribute__((address_space(1))) int gint_t;
typedef __attribute__((address_space(3))) int lint_t;

typedef __attribute__((ext_vector_type(16))) _Float16 v16h;
typedef __attribute__((ext_vector_type(8)))  float    v8f;

union AFrag { v16h v; _Float16 h[16]; unsigned int u[8]; };
union CFrag { v8f v; float f[8]; };
union H8    { _Float16 h[8]; uint64_t q[2]; };

__device__ __forceinline__ float lrelu(float x) { return x > 0.0f ? x : ALPHA_ * x; }

__device__ __forceinline__ void wait_async0() {
#if __has_builtin(__builtin_amdgcn_s_wait_asynccnt)
  __builtin_amdgcn_s_wait_asynccnt(0);
#else
  asm volatile("s_wait_asynccnt 0" ::: "memory");
#endif
}

// j-tiled f16 layout for ht:  htT[(bh*(N/32) + j/32)*DOUT + o]*32 + (j%32)
// -> WMMA B-fragments are fully contiguous 2KB blocks per wave.
__device__ __forceinline__ size_t ht_idx(int bh, int o, int n) {
  return (((size_t)bh * (N_ / 32) + (n >> 5)) * DOUT_ + o) * 32 + (n & 31);
}

// ---------------------------------------------------------------------------
// Kernel 0: f32 -> f16 conversion (x and Ww), done ONCE instead of per-tile.
// ---------------------------------------------------------------------------
__global__ __launch_bounds__(256)
void k_cvt(const float* __restrict__ src, _Float16* __restrict__ dst)
{
  size_t i = ((size_t)blockIdx.x * 256 + threadIdx.x) * 2;
  float2 f = *(const float2*)(src + i);
  dst[i]     = (_Float16)f.x;
  dst[i + 1] = (_Float16)f.y;
}

// ---------------------------------------------------------------------------
// Kernel 1: ht GEMM (f16 in, f32 acc).  ht = x@Ww^T + Wb ; htp = 1.2*acc + Wbp
// One block = 16 rows x 256 cols for one head; 8 waves, 2 WMMA per K-step.
// ---------------------------------------------------------------------------
__global__ __launch_bounds__(256)
void k_ht_gemm(const _Float16* __restrict__ xh,   // (B*N, DIN) f16
               const _Float16* __restrict__ Wwh,  // (H, DOUT, DIN) f16
               const float* __restrict__ Wb,      // (H, DOUT)
               const float* __restrict__ Wbp,     // (H, DOUT)
               _Float16* __restrict__ htT,        // j-tiled, see ht_idx
               _Float16* __restrict__ htTp)
{
  __shared__ _Float16 xs[16][32];
  const int m0  = blockIdx.x * 16;
  const int hh  = blockIdx.y;
  const int tid = threadIdx.x;
  const int w = tid >> 5, lane = tid & 31;

  CFrag acc0, acc1;
#pragma unroll
  for (int i = 0; i < 8; ++i) { acc0.f[i] = 0.0f; acc1.f[i] = 0.0f; }

  const int nb0 = w * 32;
  for (int kk = 0; kk < DIN_; kk += 32) {
    // stage 16x32 f16 tile of x into LDS (pure copy: 4B per thread)
    {
      int e0 = tid * 2;
      int r = e0 >> 5, c = e0 & 31;
      const _Float16* gsrc = xh + (size_t)(m0 + r) * DIN_ + kk + c;
#if defined(HAVE_ASYNC_B32)
      __builtin_amdgcn_global_load_async_to_lds_b32(
          (gint_t*)(void*)gsrc,
          (lint_t*)(void*)&xs[r][c],
          0, 0);
      wait_async0();
#else
      *(unsigned int*)&xs[r][c] = *(const unsigned int*)gsrc;
#endif
    }
    __syncthreads();

    // A fragment: 16x32 f16 (lane L: M=L%16; K=(L>=16?16:0)+2v+p, contiguous)
    AFrag af;
    {
      int Mr = lane & 15;
#pragma unroll
      for (int v = 0; v < 8; ++v) {
        int kb = (v * 2) + ((lane & 16) ? 16 : 0);
        af.h[2 * v]     = xs[Mr][kb];
        af.h[2 * v + 1] = xs[Mr][kb + 1];
      }
    }

#pragma unroll
    for (int nt = 0; nt < 2; ++nt) {
      // B fragment: 32x16; lane L: N=L%16, K=(L>=16?16:0)+2v+p (contiguous f16)
      AFrag bf;
      int nl = lane & 15;
      int kB = (lane & 16) ? 16 : 0;
      const unsigned int* p = (const unsigned int*)(
          Wwh + ((size_t)hh * DOUT_ + nb0 + nt * 16 + nl) * DIN_ + kk + kB);
#pragma unroll
      for (int v = 0; v < 8; ++v) bf.u[v] = p[v];
      if (nt == 0)
        acc0.v = __builtin_amdgcn_wmma_f32_16x16x32_f16(false, af.v, false, bf.v,
                                                        (short)0, acc0.v, false, false);
      else
        acc1.v = __builtin_amdgcn_wmma_f32_16x16x32_f16(false, af.v, false, bf.v,
                                                        (short)0, acc1.v, false, false);
    }
    __syncthreads();
  }

  // C layout: VGPR c, lane L: M = c + (L>=16?8:0), N = L%16.
  // For a fixed lane the 8 c-values are consecutive rows -> one 16B store.
  const int b  = m0 / N_;
  const int bh = b * H_ + hh;
  const int n0 = (m0 - b * N_) + ((lane & 16) ? 8 : 0);
#pragma unroll
  for (int nt = 0; nt < 2; ++nt) {
    int o = nb0 + nt * 16 + (lane & 15);
    float wb  = Wb[hh * DOUT_ + o];
    float wbp = Wbp[hh * DOUT_ + o];
    H8 ua, ub;
#pragma unroll
    for (int c = 0; c < 8; ++c) {
      float s = nt ? acc1.f[c] : acc0.f[c];
      ua.h[c] = (_Float16)(s + wb);
      ub.h[c] = (_Float16)((1.0f + LAM_) * s + wbp);
    }
    size_t base = ht_idx(bh, o, n0);   // 16B aligned (n0%32 in {0,8,16,24})
    uint64_t* da = (uint64_t*)(htT + base);
    uint64_t* db = (uint64_t*)(htTp + base);
    da[0] = ua.q[0]; da[1] = ua.q[1];
    db[0] = ub.q[0]; db[1] = ub.q[1];
  }
}

// ---------------------------------------------------------------------------
// Kernel 2: a1/a2 reductions for both branches. One wave per (b,h,n) row.
// ---------------------------------------------------------------------------
__global__ __launch_bounds__(256)
void k_a12(const _Float16* __restrict__ htT, const _Float16* __restrict__ htTp,
           const float* __restrict__ a1w, const float* __restrict__ a1b,
           const float* __restrict__ a2w, const float* __restrict__ a2b,
           const float* __restrict__ a1bp, const float* __restrict__ a2bp,
           float* __restrict__ a1, float* __restrict__ a2,
           float* __restrict__ a1p, float* __restrict__ a2p)
{
  const int tid = threadIdx.x;
  const int w = tid >> 5, lane = tid & 31;
  const int row = blockIdx.x * 8 + w;      // over B*H*N
  const int bh = row / N_, n = row - bh * N_;
  const int hh = bh % H_;
  const size_t rb = ht_idx(bh, 0, n);
  float s1 = 0, s2 = 0, s1p = 0, s2p = 0;
  for (int o = lane; o < DOUT_; o += 32) {
    float hv  = (float)htT [rb + (size_t)o * 32];
    float hvp = (float)htTp[rb + (size_t)o * 32];
    float w1 = a1w[hh * DOUT_ + o];
    float w2 = a2w[hh * DOUT_ + o];
    s1  += hv * w1;  s2  += hv * w2;
    s1p += hvp * w1; s2p += hvp * w2;
  }
#pragma unroll
  for (int off = 16; off >= 1; off >>= 1) {
    s1  += __shfl_xor(s1, off, 32);
    s2  += __shfl_xor(s2, off, 32);
    s1p += __shfl_xor(s1p, off, 32);
    s2p += __shfl_xor(s2p, off, 32);
  }
  if (lane == 0) {
    a1[row]  = s1 + a1b[hh];
    a2[row]  = s2 + a2b[hh];
    a1p[row] = (1.0f + LAM_) * s1p + a1bp[hh];
    a2p[row] = (1.0f + LAM_) * s2p + a2bp[hh];
  }
}

// ---------------------------------------------------------------------------
// Kernel 3: max_j a2 per (b,h)   (lrelu monotone -> m_i = lrelu(a1_i + max a2))
// ---------------------------------------------------------------------------
__global__ __launch_bounds__(256)
void k_a2max(const float* __restrict__ a2, const float* __restrict__ a2p,
             float* __restrict__ a2max, float* __restrict__ a2maxp)
{
  __shared__ float sm[256], smp[256];
  const int bh = blockIdx.x;
  const int tid = threadIdx.x;
  float m = -3.4e38f, mp = -3.4e38f;
  for (int n = tid; n < N_; n += 256) {
    m  = fmaxf(m,  a2[bh * N_ + n]);
    mp = fmaxf(mp, a2p[bh * N_ + n]);
  }
  sm[tid] = m; smp[tid] = mp;
  __syncthreads();
  for (int s = 128; s >= 1; s >>= 1) {
    if (tid < s) {
      sm[tid]  = fmaxf(sm[tid],  sm[tid + s]);
      smp[tid] = fmaxf(smp[tid], smp[tid + s]);
    }
    __syncthreads();
  }
  if (tid == 0) { a2max[bh] = sm[0]; a2maxp[bh] = smp[0]; }
}

// ---------------------------------------------------------------------------
// Kernel 4: fused attention (flash-style single pass), BOTH branches.
// Block = (b, h, 16-row tile); 8 waves; wave w owns DOUT cols [32w,32w+32).
// ---------------------------------------------------------------------------
__global__ __launch_bounds__(256)
void k_attn(const float* __restrict__ adj,
            const _Float16* __restrict__ htT, const _Float16* __restrict__ htTp,
            const float* __restrict__ a1, const float* __restrict__ a2,
            const float* __restrict__ a1p, const float* __restrict__ a2p,
            const float* __restrict__ a2max, const float* __restrict__ a2maxp,
            float* __restrict__ ssum, float* __restrict__ ssump,
            float* __restrict__ hmixH)    // (B*H, N, DOUT)
{
  __shared__ _Float16 eA[16][32];
  __shared__ _Float16 eB[16][32];
  __shared__ float rsA[16], rsB[16];

  const int i0 = blockIdx.x * 16;
  const int hh = blockIdx.y;
  const int b  = blockIdx.z;
  const int bh = b * H_ + hh;
  const int tid = threadIdx.x;
  const int w = tid >> 5, lane = tid & 31;
  const int r1 = w;                  // wave w's threads cover rows w and w+8
  const int col = tid & 31;
  const int r2 = r1 + 8;
  const int i1 = i0 + r1, i2 = i0 + r2;

  const float a1_1  = a1[bh * N_ + i1],  a1_2  = a1[bh * N_ + i2];
  const float a1p_1 = a1p[bh * N_ + i1], a1p_2 = a1p[bh * N_ + i2];
  const float m1  = lrelu(a1_1 + a2max[bh]),   m2  = lrelu(a1_2 + a2max[bh]);
  const float mp1 = lrelu(a1p_1 + a2maxp[bh]), mp2 = lrelu(a1p_2 + a2maxp[bh]);

  float psA1 = 0, psA2 = 0, psB1 = 0, psB2 = 0;
  CFrag accA0, accA1, accB0, accB1;
#pragma unroll
  for (int i = 0; i < 8; ++i) { accA0.f[i] = 0; accA1.f[i] = 0; accB0.f[i] = 0; accB1.f[i] = 0; }

  const int nbase = w * 32;
  for (int jc = 0; jc < N_; jc += 32) {
    int j = jc + col;
    float a2v  = a2[bh * N_ + j];
    float a2pv = a2p[bh * N_ + j];
    float adj1 = adj[((size_t)b * N_ + i1) * N_ + j];
    float adj2 = adj[((size_t)b * N_ + i2) * N_ + j];
    float e11 = __expf(lrelu(a1_1  + a2v)  - m1)  * adj1;
    float e21 = __expf(lrelu(a1_2  + a2v)  - m2)  * adj2;
    float f11 = __expf(lrelu(a1p_1 + a2pv) - mp1) * adj1;
    float f21 = __expf(lrelu(a1p_2 + a2pv) - mp2) * adj2;
    psA1 += e11; psA2 += e21; psB1 += f11; psB2 += f21;
    eA[r1][col] = (_Float16)e11;
    eA[r2][col] = (_Float16)e21;
    eB[r1][col] = (_Float16)f11;
    eB[r2][col] = (_Float16)f21;
    __syncthreads();

    AFrag afA, afB;
    {
      int Mr = lane & 15;
#pragma unroll
      for (int v = 0; v < 8; ++v) {
        int kb = (v * 2) + ((lane & 16) ? 16 : 0);
        afA.h[2 * v]     = eA[Mr][kb];
        afA.h[2 * v + 1] = eA[Mr][kb + 1];
        afB.h[2 * v]     = eB[Mr][kb];
        afB.h[2 * v + 1] = eB[Mr][kb + 1];
      }
    }

    const int jt = jc >> 5;
    int kB = (lane & 16) ? 16 : 0;
    int nl = lane & 15;
#pragma unroll
    for (int nt = 0; nt < 2; ++nt) {
      int o = nbase + nt * 16 + nl;
      // tiled layout: wave's fragment = one contiguous 2KB block
      const unsigned int* pA = (const unsigned int*)(
          htT  + (((size_t)bh * (N_ / 32) + jt) * DOUT_ + o) * 32 + kB);
      const unsigned int* pB = (const unsigned int*)(
          htTp + (((size_t)bh * (N_ / 32) + jt) * DOUT_ + o) * 32 + kB);
      AFrag bfA, bfB;
#pragma unroll
      for (int v = 0; v < 8; ++v) { bfA.u[v] = pA[v]; bfB.u[v] = pB[v]; }
      if (nt == 0) {
        accA0.v = __builtin_amdgcn_wmma_f32_16x16x32_f16(false, afA.v, false, bfA.v,
                                                         (short)0, accA0.v, false, false);
        accB0.v = __builtin_amdgcn_wmma_f32_16x16x32_f16(false, afB.v, false, bfB.v,
                                                         (short)0, accB0.v, false, false);
      } else {
        accA1.v = __builtin_amdgcn_wmma_f32_16x16x32_f16(false, afA.v, false, bfA.v,
                                                         (short)0, accA1.v, false, false);
        accB1.v = __builtin_amdgcn_wmma_f32_16x16x32_f16(false, afB.v, false, bfB.v,
                                                         (short)0, accB1.v, false, false);
      }
    }
    __builtin_prefetch((const void*)(adj + ((size_t)b * N_ + i1) * N_ + jc + 32), 0, 1);
    __syncthreads();
  }

#pragma unroll
  for (int off = 16; off >= 1; off >>= 1) {
    psA1 += __shfl_xor(psA1, off, 32);
    psA2 += __shfl_xor(psA2, off, 32);
    psB1 += __shfl_xor(psB1, off, 32);
    psB2 += __shfl_xor(psB2, off, 32);
  }
  if (lane == 0) { rsA[r1] = psA1; rsA[r2] = psA2; rsB[r1] = psB1; rsB[r2] = psB2; }
  __syncthreads();
  if (tid < 16) {
    float sA = rsA[tid]; sA = (sA == 0.0f) ? 1.0f : sA;
    float sB = rsB[tid]; sB = (sB == 0.0f) ? 1.0f : sB;
    rsA[tid] = sA; rsB[tid] = sB;
    ssum [bh * N_ + i0 + tid] = sA;
    ssump[bh * N_ + i0 + tid] = sB;
  }
  __syncthreads();

#pragma unroll
  for (int nt = 0; nt < 2; ++nt) {
    int o = nbase + nt * 16 + (lane & 15);
#pragma unroll
    for (int c = 0; c < 8; ++c) {
      int Mrow = c + ((lane & 16) ? 8 : 0);
      float hp  = (nt ? accA1.f[c] : accA0.f[c]) / rsA[Mrow];
      float hpp = (nt ? accB1.f[c] : accB0.f[c]) / rsB[Mrow];
      float hmix = hpp * (hp / (hpp + 1e-6f));
      hmixH[((size_t)bh * N_ + (i0 + Mrow)) * DOUT_ + o] = hmix;
    }
  }
}

// ---------------------------------------------------------------------------
// Kernel 5: out = tanh(x + mean_h hmix)
// ---------------------------------------------------------------------------
__global__ __launch_bounds__(256)
void k_out(const float* __restrict__ x, const float* __restrict__ hmixH,
           float* __restrict__ out)
{
  size_t idx = (size_t)blockIdx.x * 256 + threadIdx.x;    // < B*N*DOUT
  int o = (int)(idx % DOUT_);
  int n = (int)((idx / DOUT_) % N_);
  int b = (int)(idx / ((size_t)DOUT_ * N_));
  float s = 0.0f;
#pragma unroll
  for (int hh = 0; hh < H_; ++hh)
    s += hmixH[((size_t)(b * H_ + hh) * N_ + n) * DOUT_ + o];
  out[idx] = tanhf(x[idx] + s * (1.0f / H_));
}

// ---------------------------------------------------------------------------
// Kernel 6: att_weights = avp * (av / (avp + 1e-6)); recompute e from a1/a2
// ---------------------------------------------------------------------------
__global__ __launch_bounds__(256)
void k_attw(const float* __restrict__ adj,
            const float* __restrict__ a1, const float* __restrict__ a2,
            const float* __restrict__ a1p, const float* __restrict__ a2p,
            const float* __restrict__ a2max, const float* __restrict__ a2maxp,
            const float* __restrict__ ssum, const float* __restrict__ ssump,
            float* __restrict__ attw)     // (B,H,N,N)
{
  size_t g = (size_t)blockIdx.x * 256 + threadIdx.x;      // < B*N*N
  int j = (int)(g % N_);
  int i = (int)((g / N_) % N_);
  int b = (int)(g / ((size_t)N_ * N_));
  float adjv = adj[g];
#pragma unroll
  for (int hh = 0; hh < H_; ++hh) {
    int bh = b * H_ + hh;
    float a1v = a1[bh * N_ + i];
    float a2v = a2[bh * N_ + j];
    float m = lrelu(a1v + a2max[bh]);
    float e = __expf(lrelu(a1v + a2v) - m) * adjv;
    float av = e / ssum[bh * N_ + i];
    float a1pv = a1p[bh * N_ + i];
    float a2pv = a2p[bh * N_ + j];
    float mp = lrelu(a1pv + a2maxp[bh]);
    float ep = __expf(lrelu(a1pv + a2pv) - mp) * adjv;
    float avp = ep / ssump[bh * N_ + i];
    attw[((size_t)bh * N_ + i) * N_ + j] = avp * (av / (avp + 1e-6f));
  }
}

// ---------------------------------------------------------------------------
extern "C" void kernel_launch(void* const* d_in, const int* in_sizes, int n_in,
                              void* d_out, int out_size, void* d_ws, size_t ws_size,
                              hipStream_t stream)
{
  const float* x    = (const float*)d_in[0];   // h (B,N,1,DIN) == x (B,N,DIN)
  const float* adj  = (const float*)d_in[1];
  const float* Ww   = (const float*)d_in[2];
  const float* Wb   = (const float*)d_in[3];
  const float* a1w  = (const float*)d_in[4];
  const float* a1b  = (const float*)d_in[5];
  const float* a2w  = (const float*)d_in[6];
  const float* a2b  = (const float*)d_in[7];
  const float* Wbp  = (const float*)d_in[8];
  const float* a1bp = (const float*)d_in[9];
  const float* a2bp = (const float*)d_in[10];

  const size_t BHN = (size_t)B_ * H_ * N_;
  const size_t HT  = (size_t)B_ * H_ * DOUT_ * N_;
  const size_t XN  = (size_t)B_ * N_ * DIN_;
  const size_t WN  = (size_t)H_ * DOUT_ * DIN_;

  char* ws = (char*)d_ws;
  _Float16* htT  = (_Float16*)ws;  ws += HT * sizeof(_Float16);
  _Float16* htTp = (_Float16*)ws;  ws += HT * sizeof(_Float16);
  _Float16* xh   = (_Float16*)ws;  ws += XN * sizeof(_Float16);
  _Float16* Wwh  = (_Float16*)ws;  ws += WN * sizeof(_Float16);
  float* a1  = (float*)ws; ws += BHN * sizeof(float);
  float* a2  = (float*)ws; ws += BHN * sizeof(float);
  float* a1p = (float*)ws; ws += BHN * sizeof(float);
  float* a2p = (float*)ws; ws += BHN * sizeof(float);
  float* ss  = (float*)ws; ws += BHN * sizeof(float);
  float* ssp = (float*)ws; ws += BHN * sizeof(float);
  float* a2m  = (float*)ws; ws += 256;
  float* a2mp = (float*)ws; ws += 256;
  float* hmixH = (float*)ws;   // HT floats

  k_cvt<<<(unsigned)(XN / 512), 256, 0, stream>>>(x, xh);
  k_cvt<<<(unsigned)(WN / 512), 256, 0, stream>>>(Ww, Wwh);
  k_ht_gemm<<<dim3((B_ * N_) / 16, H_), 256, 0, stream>>>(xh, Wwh, Wb, Wbp, htT, htTp);
  k_a12<<<(unsigned)(BHN / 8), 256, 0, stream>>>(htT, htTp, a1w, a1b, a2w, a2b,
                                                 a1bp, a2bp, a1, a2, a1p, a2p);
  k_a2max<<<B_ * H_, 256, 0, stream>>>(a2, a2p, a2m, a2mp);
  k_attn<<<dim3(N_ / 16, H_, B_), 256, 0, stream>>>(adj, htT, htTp, a1, a2, a1p, a2p,
                                                    a2m, a2mp, ss, ssp, hmixH);
  float* out = (float*)d_out;
  k_out<<<(unsigned)(((size_t)B_ * N_ * DOUT_) / 256), 256, 0, stream>>>(x, hmixH, out);
  k_attw<<<(unsigned)(((size_t)B_ * N_ * N_) / 256), 256, 0, stream>>>(
      adj, a1, a2, a1p, a2p, a2m, a2mp, ss, ssp, out + (size_t)B_ * N_ * DOUT_);

  (void)in_sizes; (void)n_in; (void)out_size; (void)ws_size;
}
